// LSTM_39642548142415
// MI455X (gfx1250) — compile-verified, matrix operand order checked by reading, not run on hardware
//
#include <hip/hip_runtime.h>
#include <hip/hip_bf16.h>

// ---------------------------------------------------------------------------
// LSTM for MI455X (gfx1250, wave32).
//  - bf16 WMMA (v_wmma_f32_16x16x32_bf16), f32 accumulate.
//  - All 48 weight B-fragments per wave live in registers (384 VGPRs via
//    s_set_vgpr_msb high banks). Weights touch L2 once.
//  - A-side fragments loaded in 16-fragment batches; an asm scheduling fence
//    (memory clobber + acc passthrough) stops the scheduler from
//    re-serializing load->wait->wmma, so each batch pays ONE L2 latency.
//  - x-contribution computed BEFORE the device barrier wait (independent of
//    h), hiding barrier latency.
//  - h double-buffered (parity by t); per-step counters (no generation wrap).
// ---------------------------------------------------------------------------

typedef __attribute__((ext_vector_type(16))) __bf16 v16bf;
typedef __attribute__((ext_vector_type(8)))  __bf16 v8bf;
typedef __attribute__((ext_vector_type(8)))  float  v8f;

#define LSTM_T     512
#define LSTM_B     64
#define LSTM_DIN   512
#define LSTM_DLAT  1024
#define NWG        256   // 4 row-tiles (batch) x 64 col-tiles (h)
#define TPB        128   // 4 waves of 32

static __device__ __forceinline__ unsigned short f32_to_bf16_rne(float f) {
  union { float f; unsigned int u; } v; v.f = f;
  unsigned int u = v.u;
  u += 0x7fffu + ((u >> 16) & 1u);   // round-to-nearest-even
  return (unsigned short)(u >> 16);
}

static __device__ __forceinline__ v16bf load_frag(const unsigned short* p) {
  // Two contiguous 16-byte chunks -> one 16-element bf16 fragment
  // (16-bit A/B WMMA VGPR layout: K elems 0..7, then 8..15 at +16).
  union { v16bf v; v8bf h[2]; } u;
  u.h[0] = *(const v8bf*)(p);
  u.h[1] = *(const v8bf*)(p + 16);
  return u.v;
}

// Scheduling fence: loads issued before this cannot sink below it (memory
// clobber), and WMMAs consuming `acc` cannot hoist above it (data dep).
static __device__ __forceinline__ void sched_fence(v8f& acc) {
  asm volatile("" : "+v"(acc) :: "memory");
}

static __device__ __forceinline__ float sigmoidf_dev(float x) {
  return 1.0f / (1.0f + __expf(-x));
}

// --------------------------- prep kernels ----------------------------------

__global__ void lstm_convert_x(const float* __restrict__ x,
                               unsigned short* __restrict__ xb, int n) {
  int i = blockIdx.x * blockDim.x + threadIdx.x;
  if (i < n) xb[i] = f32_to_bf16_rne(x[i]);
}

// WT[g][n][kz] = W_g[kz][n]  (bf16, N-major so B-fragments load contiguously)
__global__ void lstm_transpose_w(const float* __restrict__ Wf,
                                 const float* __restrict__ Wi,
                                 const float* __restrict__ Wo,
                                 const float* __restrict__ Wu,
                                 unsigned short* __restrict__ WT) {
  const int KZ = LSTM_DIN + LSTM_DLAT;         // 1536
  size_t idx = (size_t)blockIdx.x * blockDim.x + threadIdx.x;
  size_t total = (size_t)4 * LSTM_DLAT * KZ;   // 6,291,456
  if (idx >= total) return;
  int kz = (int)(idx % KZ);
  size_t rem = idx / KZ;
  int n = (int)(rem % LSTM_DLAT);
  int g = (int)(rem / LSTM_DLAT);
  const float* W = (g == 0) ? Wf : (g == 1) ? Wi : (g == 2) ? Wo : Wu;
  WT[idx] = f32_to_bf16_rne(W[(size_t)kz * LSTM_DLAT + n]);
}

__global__ void lstm_init_state(unsigned short* __restrict__ hb0,
                                unsigned short* __restrict__ hb1,
                                int* __restrict__ bar) {
  int i = blockIdx.x * blockDim.x + threadIdx.x;
  if (i < LSTM_B * LSTM_DLAT) { hb0[i] = 0; hb1[i] = 0; }  // h_{-1} = 0
  if (i < LSTM_T) bar[i] = 0;                              // step counters
}

// --------------------------- persistent LSTM -------------------------------

__global__ __launch_bounds__(TPB) void lstm_persist(
    const unsigned short* __restrict__ xb,   // [T][B][DIN]     bf16
    const unsigned short* __restrict__ WT,   // [4][DLAT][1536] bf16
    const float* __restrict__ bf_, const float* __restrict__ bi_,
    const float* __restrict__ bo_, const float* __restrict__ bu_,
    unsigned short* __restrict__ hb0,        // [B][DLAT] bf16, parity 0
    unsigned short* __restrict__ hb1,        // [B][DLAT] bf16, parity 1
    int* __restrict__ bar,                   // [T]
    float* __restrict__ out) {               // [T][B][DLAT] f32
  __shared__ float gateT[4][256];            // f,i,o,u 16x16 tiles
  __shared__ float cT[256];                  // persistent cell-state tile

  const int tid  = threadIdx.x;
  const int lane = tid & 31;
  const int g    = tid >> 5;                 // 0=f 1=i 2=o 3=u
  const int rowTile = blockIdx.x & 3;        // batch rows /16
  const int colTile = blockIdx.x >> 2;       // h cols    /16
  const int mrow  = lane & 15;               // A row / B col / C col index
  const int khalf = (lane >> 4) << 3;        // 0 or 8 (K half select)

  for (int e = tid; e < 256; e += TPB) cT[e] = 0.0f;

  const float* bptr = (g == 0) ? bf_ : (g == 1) ? bi_ : (g == 2) ? bo_ : bu_;
  const float bias = bptr[colTile * 16 + mrow];

  // -------- Preload ALL 48 weight B-fragments into registers (384 VGPRs).
  // B col = mrow; fragment i covers K = i*32..i*32+31 of the 1536-deep row.
  const unsigned short* Bbase =
      WT + ((size_t)g * LSTM_DLAT + (size_t)colTile * 16 + mrow) * 1536;
  v16bf Bfrag[48];
#pragma unroll
  for (int i = 0; i < 48; ++i) Bfrag[i] = load_frag(Bbase + i * 32 + khalf);

  const size_t rowoff = ((size_t)rowTile * 16 + mrow) * LSTM_DLAT;

  __syncthreads();

  for (int t = 0; t < LSTM_T; ++t) {
    const unsigned short* __restrict__ hbR = (t & 1) ? hb0 : hb1;  // h_{t-1}
    unsigned short* __restrict__ hbW       = (t & 1) ? hb1 : hb0;  // h_t

    const unsigned short* Axbase =
        xb + ((size_t)t * LSTM_B + rowTile * 16 + mrow) * LSTM_DIN;

    v8f acc;
#pragma unroll
    for (int i = 0; i < 8; ++i) acc[i] = bias;   // C col = lane%16 for all i

    // ---- x contribution (independent of h): overlaps the device barrier.
    // 32 b128 loads burst out, fence, then 16 WMMAs drain with partial waits.
    v16bf A0[16], A1[16];
#pragma unroll
    for (int i = 0; i < 16; ++i) A0[i] = load_frag(Axbase + i * 32 + khalf);
    sched_fence(acc);
#pragma unroll
    for (int i = 0; i < 16; ++i)
      acc = __builtin_amdgcn_wmma_f32_16x16x32_bf16(
          false, A0[i], false, Bfrag[i], (short)0, acc, false, false);

    // ---- wait for all WGs to have published h_{t-1}
    if (t > 0) {
      if (tid == 0) {
        while (__hip_atomic_load(&bar[t - 1], __ATOMIC_ACQUIRE,
                                 __HIP_MEMORY_SCOPE_AGENT) < NWG) {
          __builtin_amdgcn_s_sleep(1);
        }
      }
      __syncthreads();
      __threadfence();   // acquire for all lanes: no h-loads hoist above this
    }

    // ---- h contribution: K=1024 in two 16-fragment register batches.
    // Batch2 loads sit between fences -> overlap batch1 WMMAs.
    const unsigned short* Ahbase = hbR + rowoff;
#pragma unroll
    for (int i = 0; i < 16; ++i) A0[i] = load_frag(Ahbase + i * 32 + khalf);
    sched_fence(acc);
#pragma unroll
    for (int i = 0; i < 16; ++i)
      A1[i] = load_frag(Ahbase + 512 + i * 32 + khalf);
#pragma unroll
    for (int i = 0; i < 16; ++i)
      acc = __builtin_amdgcn_wmma_f32_16x16x32_bf16(
          false, A0[i], false, Bfrag[16 + i], (short)0, acc, false, false);
    sched_fence(acc);
#pragma unroll
    for (int i = 0; i < 16; ++i)
      acc = __builtin_amdgcn_wmma_f32_16x16x32_bf16(
          false, A1[i], false, Bfrag[32 + i], (short)0, acc, false, false);

    // C/D layout: lane L holds N=L%16; VGPR i holds M = i + (L/16)*8
#pragma unroll
    for (int i = 0; i < 8; ++i) {
      int m = i + ((lane >> 4) << 3);
      gateT[g][m * 16 + mrow] = acc[i];
    }

    // Prefetch next step's x tile while gates settle (global_prefetch_b8).
    if (t + 1 < LSTM_T) {
      const unsigned short* nx =
          xb + ((size_t)(t + 1) * LSTM_B + rowTile * 16) * LSTM_DIN + tid * 64;
      __builtin_prefetch(nx, 0, 1);
    }
    __syncthreads();

    // Elementwise gate fusion: 256 elems / 128 threads
    for (int e = tid; e < 256; e += TPB) {
      float fv = sigmoidf_dev(gateT[0][e]);
      float iv = sigmoidf_dev(gateT[1][e]);
      float ov = sigmoidf_dev(gateT[2][e]);
      float uv = tanhf(gateT[3][e]);
      float c  = fv * cT[e] + iv * uv;
      cT[e]    = c;
      float h  = ov * tanhf(c);
      int m = e >> 4, n = e & 15;
      int row = rowTile * 16 + m;
      int col = colTile * 16 + n;
      out[((size_t)t * LSTM_B + row) * LSTM_DLAT + col] = h;
      hbW[(size_t)row * LSTM_DLAT + col] = f32_to_bf16_rne(h);
    }

    // Publish h_t (release), no wait here: next iter's x-part hides latency.
    __threadfence();
    __syncthreads();
    if (tid == 0) {
      __hip_atomic_fetch_add(&bar[t], 1, __ATOMIC_RELEASE,
                             __HIP_MEMORY_SCOPE_AGENT);
    }
  }
}

// ------------------------------ launch -------------------------------------

extern "C" void kernel_launch(void* const* d_in, const int* in_sizes, int n_in,
                              void* d_out, int out_size, void* d_ws,
                              size_t ws_size, hipStream_t stream) {
  (void)in_sizes; (void)n_in; (void)out_size; (void)ws_size;
  const float* x  = (const float*)d_in[0];
  const float* Wf = (const float*)d_in[1];
  const float* Wi = (const float*)d_in[2];
  const float* Wo = (const float*)d_in[3];
  const float* Wu = (const float*)d_in[4];
  const float* bf = (const float*)d_in[5];
  const float* bi = (const float*)d_in[6];
  const float* bo = (const float*)d_in[7];
  const float* bu = (const float*)d_in[8];
  float* out = (float*)d_out;

  // Workspace layout (all 256B-aligned):
  const size_t xb_elems = (size_t)LSTM_T * LSTM_B * LSTM_DIN;   // 16,777,216
  const size_t wt_elems = (size_t)4 * LSTM_DLAT * 1536;         //  6,291,456
  const size_t hb_elems = (size_t)LSTM_B * LSTM_DLAT;           //     65,536
  char* ws = (char*)d_ws;
  unsigned short* xb  = (unsigned short*)ws;
  unsigned short* WT  = (unsigned short*)(ws + xb_elems * 2);
  unsigned short* hb0 = (unsigned short*)(ws + (xb_elems + wt_elems) * 2);
  unsigned short* hb1 = (unsigned short*)(ws + (xb_elems + wt_elems + hb_elems) * 2);
  int* bar = (int*)(ws + (xb_elems + wt_elems + 2 * hb_elems) * 2);

  int nx = (int)xb_elems;
  lstm_convert_x<<<(nx + 255) / 256, 256, 0, stream>>>(x, xb, nx);

  int nw = (int)wt_elems;
  lstm_transpose_w<<<(nw + 255) / 256, 256, 0, stream>>>(Wf, Wi, Wo, Wu, WT);

  lstm_init_state<<<(LSTM_B * LSTM_DLAT + 255) / 256, 256, 0, stream>>>(hb0, hb1, bar);

  lstm_persist<<<NWG, TPB, 0, stream>>>(xb, WT, bf, bi, bo, bu,
                                        hb0, hb1, bar, out);
}
